// GraphConvolution_64424509440236
// MI455X (gfx1250) — matrix operand order repaired and checked by reading, not compiled, not run on hardware
//
#include <hip/hip_runtime.h>

// GraphConvolution = relu( A(N x 512) @ kernel(512 x 128) ), A densified from COO.
// A lives in d_ws as f16, pre-permuted into WMMA A-fragment order:
//   flat = tile*8192 + (kb*32 + lane)*16 + e      (f16 elements, 16 KB per 16-row tile)
// so the GEMM stages tiles with contiguous async global->LDS copies.

typedef __attribute__((ext_vector_type(16))) _Float16 v16h;
typedef __attribute__((ext_vector_type(8)))  float    v8f;

#define F_DIM 512
#define D_DIM 128
#define KB_STEPS (F_DIM / 32)                 // 16 WMMA K-steps
#define B_FRAG_ELEMS (F_DIM * D_DIM)          // 65536 f16 = 128 KB
#define A_TILE_ELEMS (KB_STEPS * 32 * 16)     // 8192 f16  = 16 KB per tile
#define A_TILE_BYTES (A_TILE_ELEMS * 2)

// ---------------- Kernel 1: zero the f16 scratch matrix ----------------
__global__ void gc_zero(float4* __restrict__ p, long long n_vec4) {
    long long i = (long long)blockIdx.x * blockDim.x + threadIdx.x;
    long long stride = (long long)gridDim.x * blockDim.x;
    float4 z = make_float4(0.f, 0.f, 0.f, 0.f);
    for (long long k = i; k < n_vec4; k += stride) p[k] = z;
}

// ------- Kernel 2: COO scatter into f16 fragment-ordered A (pk atomics) -------
__global__ void gc_scatter(const int* __restrict__ indices,   // [NNZ,2] int32
                           const float* __restrict__ values,  // [NNZ]
                           _Float16* __restrict__ a,          // fragment-ordered
                           int nnz, int n_rows) {
    int i = blockIdx.x * blockDim.x + threadIdx.x;
    if (i >= nnz) return;
    int row = indices[2 * i + 0];
    int col = indices[2 * i + 1];
    if ((unsigned)row >= (unsigned)n_rows || (unsigned)col >= (unsigned)F_DIM) return;

    // inverse of the A 16x32 f16 fragment layout (ISA 7.12.2)
    int tile = row >> 4, m = row & 15;
    int kb   = col >> 5, k32 = col & 31;
    int lnHi = (k32 >> 3) & 1;
    int e    = (k32 & 7) + ((k32 & 16) >> 1);
    int lane = m + (lnHi << 4);
    unsigned long long flat =
        (unsigned long long)tile * A_TILE_ELEMS + (unsigned)((kb * 32 + lane) * 16 + e);

    _Float16 hv = (_Float16)values[i];
    unsigned short bits = __builtin_bit_cast(unsigned short, hv);
    unsigned int data = (unsigned int)bits << (16u * (unsigned)(flat & 1ull));
    unsigned long long gaddr =
        (unsigned long long)(uintptr_t)a + (flat & ~1ull) * 2ull;

    // packed f16 atomic add (no return) at 4-byte granularity
    asm volatile("global_atomic_pk_add_f16 %0, %1, off"
                 :: "v"(gaddr), "v"(data) : "memory");
}

// ---------------- Kernel 3: persistent WMMA GEMM + ReLU ----------------
// Block = 256 threads = 8 waves; wave w owns output cols [w*16, w*16+16).
// LDS: Bstage 128 KB (staged once) + A double buffer 2x16 KB, async-copied.
__device__ __forceinline__ void issue_tile_copy(_Float16* ldsDst,
                                                const _Float16* gSrc, int tid) {
    unsigned int lds = (unsigned int)(uintptr_t)ldsDst + (unsigned)tid * 16u;
    unsigned long long g =
        (unsigned long long)(uintptr_t)gSrc + (unsigned long long)tid * 16ull;
#pragma unroll
    for (int r = 0; r < 4; ++r) {   // 256 threads * 16 B * 4 = 16 KB
        asm volatile("global_load_async_to_lds_b128 %0, %1, off"
                     :: "v"(lds + (unsigned)(r * 4096)),
                        "v"(g + (unsigned long long)(r * 4096)) : "memory");
    }
}

__global__ void gc_wmma_gemm(const _Float16* __restrict__ aFrag, // [n_tiles][8192]
                             const float* __restrict__ kernelW,  // [512,128] f32
                             float* __restrict__ out,            // [N,128]
                             int n_rows, int n_tiles) {
    extern __shared__ char smem[];
    _Float16* Bstage = (_Float16*)smem;                          // 65536 halves
    _Float16* Abuf   = (_Float16*)(smem + B_FRAG_ELEMS * 2);     // 2 x 8192 halves

    const int tid  = threadIdx.x;
    const int lane = tid & 31;
    const int wv   = tid >> 5;

    // ---- Stage kernel matrix into LDS in B-fragment order (once) ----
    // B 32x16 f16: lane L holds col N=L%16; element e -> K = (e&7) + (e>=8 ? 16 : 0) + (L>=16 ? 8 : 0)
    for (int f = tid; f < B_FRAG_ELEMS; f += 256) {
        int e  = f & 15;
        int ln = (f >> 4) & 31;
        int w  = (f >> 9) & 7;
        int kb = f >> 12;
        int K  = kb * 32 + (e & 7) + ((e & 8) << 1) + ((ln >= 16) ? 8 : 0);
        int Nc = w * 16 + (ln & 15);
        Bstage[f] = (_Float16)kernelW[K * D_DIM + Nc];
    }

    const int stride = gridDim.x;
    int tile = blockIdx.x;

    if (tile < n_tiles)   // prologue: async-copy first tile into buffer 0
        issue_tile_copy(Abuf, aFrag + (size_t)tile * A_TILE_ELEMS, tid);
    __syncthreads();      // B staged

    int buf = 0;
    for (; tile < n_tiles; tile += stride) {
        int nxt = tile + stride;                     // block-uniform
        if (nxt < n_tiles) {
            issue_tile_copy(Abuf + (buf ^ 1) * A_TILE_ELEMS,
                            aFrag + (size_t)nxt * A_TILE_ELEMS, tid);
            asm volatile("s_wait_asynccnt 0x4" ::: "memory"); // current tile done
        } else {
            asm volatile("s_wait_asynccnt 0x0" ::: "memory");
        }
        __syncthreads();  // every wave's copies for this tile are in LDS

        const _Float16* As = Abuf + buf * A_TILE_ELEMS;
        v8f c = {};
#pragma unroll
        for (int kb = 0; kb < KB_STEPS; ++kb) {
            v16h av = *(const v16h*)(As + (kb * 32 + lane) * 16);
            v16h bv = *(const v16h*)(Bstage + ((kb * 8 + wv) * 32 + lane) * 16);
            c = __builtin_amdgcn_wmma_f32_16x16x32_f16(
                    false, av, false, bv, (short)0, c, false, false);
        }

        // Epilogue: ReLU + store per C/D layout (VGPR r: M=r / r+8)
        {
            const int rowBase = tile * 16;
            int col  = wv * 16 + (lane & 15);
            int mOff = (lane >= 16) ? 8 : 0;
#pragma unroll
            for (int r = 0; r < 8; ++r) {
                int row = rowBase + r + mOff;
                if (row < n_rows)
                    out[(long long)row * D_DIM + col] = fmaxf(c[r], 0.f);
            }
        }
        __syncthreads();  // all reads of this buffer done before it is refilled
        buf ^= 1;
    }
}

extern "C" void kernel_launch(void* const* d_in, const int* in_sizes, int n_in,
                              void* d_out, int out_size, void* d_ws, size_t ws_size,
                              hipStream_t stream) {
    // Inputs: x[N,512] f32 (unused), indices[NNZ,2] i32, values[NNZ] f32,
    //         kernel[512,128] f32
    const int*   indices = (const int*)d_in[1];
    const float* values  = (const float*)d_in[2];
    const float* kernelW = (const float*)d_in[3];

    const int n_rows  = in_sizes[0] / F_DIM;      // N = 100000
    const int nnz     = in_sizes[2];              // 1.6M
    const int n_tiles = (n_rows + 15) / 16;       // 6250

    _Float16* A  = (_Float16*)d_ws;               // n_tiles*8192 f16 (102.4 MB)
    float*   out = (float*)d_out;

    // 1) zero A (f16 region, as float4 vectors)
    const long long a_bytes = (long long)n_tiles * A_TILE_BYTES;
    gc_zero<<<4096, 256, 0, stream>>>((float4*)A, a_bytes / 16);

    // 2) scatter COO -> fragment-ordered f16 A (pk_add_f16 atomics)
    gc_scatter<<<(nnz + 255) / 256, 256, 0, stream>>>(indices, values, A, nnz, n_rows);

    // 3) WMMA GEMM + ReLU (async double-buffered A tiles; 160 KB dynamic LDS)
    const size_t shmem = (size_t)B_FRAG_ELEMS * 2 + 2 * A_TILE_BYTES;
    gc_wmma_gemm<<<640, 256, shmem, stream>>>(A, kernelW, out, n_rows, n_tiles);
}
